// PatchNet_79800492360088
// MI455X (gfx1250) — matrix operand order repaired and checked by reading, not compiled, not run on hardware
//
#include <hip/hip_runtime.h>
#include <hip/hip_bf16.h>
#include <math.h>

typedef _Float16 v8h  __attribute__((ext_vector_type(8)));
typedef _Float16 v16h __attribute__((ext_vector_type(16)));
typedef float    v8f  __attribute__((ext_vector_type(8)));

#define NB      4          // B
#define NPTS    16384
#define NNODES  32
#define KSEL    1024
#define PSEL    2048       // 2*KSEL
#define HID     256
#define LDIM    128
#define MROWS   (NB*PSEL*NNODES)   // 262144

// ---- d_out layout (floats, return-order concatenation) ----
#define O_WSDF  0
#define O_VIS   8192
#define O_SEL   1581056
#define O_CI    3153920
#define O_PW    3940352
#define O_SDFS  4202496
#define O_SD    4464640
#define O_EXT   4726784
#define O_CEN   4726912

// ---- workspace layout (bytes) ----
// wsP floats: [0,128) const, [128,512) scales, [512,1664) R, [1664,2048) centers
#define WS_PARAMS   0
#define WS_IDX      8192
#define WS_HMOD     532480
#define WS_WT       1318912
#define WS_BIAS     1974272

// ---- k_siren dynamic LDS layout (bytes) ----
#define SM_X0    0         // 128x256 f16 = 65536
#define SM_X1    65536     // 128x256 f16 = 65536
#define SM_W     131072    // 256x256 f16 = 131072 (per-layer weights, n-major)
#define SM_HM    262144    // 32x256  f32 = 32768
#define SM_BS    294912    // 256     f32 = 1024
#define SM_CI    295936    // 128x3   f32 = 1536
#define SM_TOTAL 297472    // <= 320KB WGP LDS

struct ModPtrs { const float* W[6]; const float* bvec[6]; };
struct WPtrs   { const float* W[5]; const float* bvec[6]; };

// ------------------------------------------------------------------
// 1) node parameters
// ------------------------------------------------------------------
__global__ void k_params(const float* __restrict__ ext,
                         float* __restrict__ wsP,
                         float* __restrict__ out_ext,
                         float* __restrict__ out_cen)
{
    int i = threadIdx.x;
    if (i >= NB * NNODES) return;
    const float* e = ext + (size_t)i * 11;
    float c = e[0];
    wsP[i] = (c > 20.f) ? c : log1pf(expf(c));          // softplus
    float s0 = expf(e[1]), s1 = expf(e[2]), s2 = expf(e[3]);
    wsP[128 + i*3 + 0] = s0;
    wsP[128 + i*3 + 1] = s1;
    wsP[128 + i*3 + 2] = s2;
    float w = e[4], x = e[5], y = e[6], z = e[7];
    float nq = sqrtf(w*w + x*x + y*y + z*z) + 1e-8f;
    w /= nq; x /= nq; y /= nq; z /= nq;
    float* R = wsP + 512 + i*9;
    R[0] = 1.f - 2.f*(y*y + z*z); R[1] = 2.f*(x*y - w*z); R[2] = 2.f*(x*z + w*y);
    R[3] = 2.f*(x*y + w*z); R[4] = 1.f - 2.f*(x*x + z*z); R[5] = 2.f*(y*z - w*x);
    R[6] = 2.f*(x*z - w*y); R[7] = 2.f*(y*z + w*x); R[8] = 1.f - 2.f*(x*x + y*y);
    wsP[1664 + i*3 + 0] = e[8];
    wsP[1664 + i*3 + 1] = e[9];
    wsP[1664 + i*3 + 2] = e[10];
    out_ext[i] = s0;
    out_cen[i*3 + 0] = e[8];
    out_cen[i*3 + 1] = e[9];
    out_cen[i*3 + 2] = e[10];
}

// ------------------------------------------------------------------
// 2) weight prep: f16 transpose of siren W1..W5, gather biases
// ------------------------------------------------------------------
__global__ void k_prep(WPtrs wp, _Float16* __restrict__ WT, float* __restrict__ BiasAll)
{
    int idx = blockIdx.x * 256 + threadIdx.x;
    if (idx < 5 * 65536) {
        int l = idx >> 16;
        int n = (idx >> 8) & 255;
        int k = idx & 255;
        WT[idx] = (_Float16)wp.W[l][(size_t)k * HID + n];   // WT[l][n][k] = W[l+1][k][n]
    }
    if (idx < 6 * 256) {
        BiasAll[idx] = wp.bvec[idx >> 8][idx & 255];
    }
}

// ------------------------------------------------------------------
// 3) modulation network (tiny): Hmod[6][b*32+n][256]
// ------------------------------------------------------------------
__global__ __launch_bounds__(256)
void k_mod(const float* __restrict__ zc, ModPtrs mp, float* __restrict__ Hmod)
{
    __shared__ float zsh[LDIM];
    __shared__ float hA[HID], hB[HID];
    int bn = blockIdx.x;
    int c  = threadIdx.x;
    if (c < LDIM) zsh[c] = zc[(size_t)bn * LDIM + c];
    __syncthreads();

    float s = mp.bvec[0][c];
    for (int j = 0; j < LDIM; ++j) s += zsh[j] * mp.W[0][(size_t)j * HID + c];
    float h = fmaxf(s, 0.f);
    hA[c] = h;
    Hmod[(size_t)bn * HID + c] = h;
    __syncthreads();

    float* hc = hA; float* hn = hB;
    for (int L = 1; L < 6; ++L) {
        const float* W = mp.W[L];
        float s2 = mp.bvec[L][c];
        for (int j = 0; j < HID;  ++j) s2 += hc[j]  * W[(size_t)j * HID + c];
        for (int j = 0; j < LDIM; ++j) s2 += zsh[j] * W[(size_t)(HID + j) * HID + c];
        float hv = fmaxf(s2, 0.f);
        hn[c] = hv;
        Hmod[(size_t)L * (NB*NNODES*HID) + (size_t)bn * HID + c] = hv;
        __syncthreads();
        float* t = hc; hc = hn; hn = t;
    }
}

// ------------------------------------------------------------------
// 4) top-k via full in-LDS bitonic sort (128 KB tile per workgroup)
// ------------------------------------------------------------------
__global__ __launch_bounds__(1024)
void k_topk(const float* __restrict__ pts, const float* __restrict__ prob,
            const float* __restrict__ wsP, int* __restrict__ idxsel)
{
    extern __shared__ char smem[];
    unsigned long long* keys = (unsigned long long*)smem;
    const int bn = blockIdx.x, b = bn >> 5, n = bn & 31;
    const int tid = threadIdx.x;
    const float* R  = wsP + 512  + bn * 9;
    const float* sc = wsP + 128  + bn * 3;
    const float* ce = wsP + 1664 + bn * 3;
    float R0=R[0],R1=R[1],R2=R[2],R3=R[3],R4=R[4],R5=R[5],R6=R[6],R7=R[7],R8=R[8];
    float s0=sc[0], s1=sc[1], s2=sc[2], c0=ce[0], c1=ce[1], c2=ce[2];

    for (int p = tid; p < NPTS; p += 1024) {
        const float* pt = pts + ((size_t)b * NPTS + p) * 6;
        float d0 = pt[0]-c0, d1 = pt[1]-c1, d2 = pt[2]-c2;
        float l0 = (R0*d0 + R3*d1 + R6*d2) / s0;      // R^T diff / scales
        float l1 = (R1*d0 + R4*d1 + R7*d2) / s1;
        float l2 = (R2*d0 + R5*d1 + R8*d2) / s2;
        float dist = sqrtf(l0*l0 + l1*l1 + l2*l2);
        float key = (dist < 1.0f) ? prob[((size_t)b*NPTS + p)*NNODES + n] : -dist;
        unsigned u = __float_as_uint(key);
        u = (u & 0x80000000u) ? ~u : (u | 0x80000000u);   // order-preserving map
        keys[p] = ((unsigned long long)u << 32) | (unsigned)(~p); // ~p: stable ties
    }
    __syncthreads();

    for (unsigned k = 2; k <= NPTS; k <<= 1) {
        for (unsigned j = k >> 1; j > 0; j >>= 1) {
            for (unsigned i = tid; i < NPTS; i += 1024) {
                unsigned ixj = i ^ j;
                if (ixj > i) {
                    unsigned long long a = keys[i], bb = keys[ixj];
                    bool desc = ((i & k) == 0);
                    if (desc ? (a < bb) : (a > bb)) { keys[i] = bb; keys[ixj] = a; }
                }
            }
            __syncthreads();
        }
    }
    if (tid < KSEL)
        idxsel[(size_t)bn * KSEL + tid] = (int)(~(unsigned)(keys[tid] & 0xFFFFFFFFull));
}

// ------------------------------------------------------------------
// 5) gather selected points + off-surface sphere; write sel/vis/ci
// ------------------------------------------------------------------
__global__ __launch_bounds__(256)
void k_gather(const float* __restrict__ pts, const float* __restrict__ odir,
              const float* __restrict__ ou, const int* __restrict__ idxsel,
              const float* __restrict__ wsP,
              float* __restrict__ out_vis, float* __restrict__ out_sel,
              float* __restrict__ out_ci)
{
    int t = blockIdx.x * 256 + threadIdx.x;       // [b][ps][n]
    int n  = t & 31;
    int ps = (t >> 5) & (PSEL - 1);
    int b  = t >> 16;
    int bn = b * NNODES + n;
    const float* R  = wsP + 512  + bn * 9;
    const float* sc = wsP + 128  + bn * 3;
    const float* ce = wsP + 1664 + bn * 3;

    float lp0, lp1, lp2, ln0, ln1, ln2;
    if (ps < KSEL) {
        int p = idxsel[(size_t)bn * KSEL + ps];
        const float* pt = pts + ((size_t)b * NPTS + p) * 6;
        float d0 = pt[0]-ce[0], d1 = pt[1]-ce[1], d2 = pt[2]-ce[2];
        lp0 = (R[0]*d0 + R[3]*d1 + R[6]*d2) / sc[0];
        lp1 = (R[1]*d0 + R[4]*d1 + R[7]*d2) / sc[1];
        lp2 = (R[2]*d0 + R[5]*d1 + R[8]*d2) / sc[2];
        ln0 = R[0]*pt[3] + R[3]*pt[4] + R[6]*pt[5];
        ln1 = R[1]*pt[3] + R[4]*pt[4] + R[7]*pt[5];
        ln2 = R[2]*pt[3] + R[5]*pt[4] + R[8]*pt[5];
    } else {
        int k2 = ps - KSEL;
        const float* dv = odir + ((size_t)b * KSEL + k2) * 3;
        float u = ou[(size_t)b * KSEL + k2];
        float nn = sqrtf(dv[0]*dv[0] + dv[1]*dv[1] + dv[2]*dv[2]) + 1e-8f;
        float rr = cbrtf(u);
        lp0 = dv[0]/nn*rr; lp1 = dv[1]/nn*rr; lp2 = dv[2]/nn*rr;
        ln0 = lp0; ln1 = lp1; ln2 = lp2;
    }
    size_t o6 = (size_t)t * 6;
    out_sel[o6+0]=lp0; out_sel[o6+1]=lp1; out_sel[o6+2]=lp2;
    out_sel[o6+3]=ln0; out_sel[o6+4]=ln1; out_sel[o6+5]=ln2;

    float p0 = lp0*sc[0], p1 = lp1*sc[1], p2 = lp2*sc[2];
    out_vis[o6+0] = R[0]*p0 + R[1]*p1 + R[2]*p2 + ce[0];
    out_vis[o6+1] = R[3]*p0 + R[4]*p1 + R[5]*p2 + ce[1];
    out_vis[o6+2] = R[6]*p0 + R[7]*p1 + R[8]*p2 + ce[2];
    out_vis[o6+3] = R[0]*ln0 + R[1]*ln1 + R[2]*ln2;
    out_vis[o6+4] = R[3]*ln0 + R[4]*ln1 + R[5]*ln2;
    out_vis[o6+5] = R[6]*ln0 + R[7]*ln1 + R[8]*ln2;

    size_t o3 = (size_t)t * 3;
    out_ci[o3+0] = lp0; out_ci[o3+1] = lp1; out_ci[o3+2] = lp2;
}

// ------------------------------------------------------------------
// 6) fused SIREN decoder: 128-row activation tiles + per-layer weight
//    matrix all resident in LDS (290 KB of the 320 KB WGP LDS),
//    WMMA f16->f32. B-fragments now come from LDS (8x less L2 traffic).
// ------------------------------------------------------------------
__global__ __launch_bounds__(512)
void k_siren(const float* __restrict__ CIg, const float* __restrict__ W0,
             const _Float16* __restrict__ WT, const float* __restrict__ BiasAll,
             const float* __restrict__ Hmod, const float* __restrict__ W6,
             const float* __restrict__ b6, float* __restrict__ sdf_out)
{
    extern __shared__ char smem[];
    _Float16* Xbuf0 = (_Float16*)(smem + SM_X0);   // 128x256
    _Float16* Xbuf1 = (_Float16*)(smem + SM_X1);   // 128x256
    _Float16* WL    = (_Float16*)(smem + SM_W);    // 256(n) x 256(k)
    float* Hm = (float*)(smem + SM_HM);            // 32x256
    float* Bs = (float*)(smem + SM_BS);            // 256
    float* CI = (float*)(smem + SM_CI);            // 128x3

    const int tid  = threadIdx.x;
    const int row0 = blockIdx.x * 128;
    const int b    = row0 >> 16;                   // rows per batch = 65536

    for (int i = tid; i < 128*3; i += 512) CI[i] = CIg[(size_t)row0*3 + i];
    const float* H0 = Hmod + (size_t)b * (NNODES*HID);
    for (int i = tid; i < NNODES*HID; i += 512) Hm[i] = H0[i];
    for (int i = tid; i < HID; i += 512) Bs[i] = BiasAll[i];
    __syncthreads();

    // ---- layer 0 (din=3, scalar) ----
    for (int i = tid; i < 128*HID; i += 512) {
        int r = i >> 8, c = i & 255;
        float x = Bs[c] + CI[r*3+0]*W0[c] + CI[r*3+1]*W0[HID+c] + CI[r*3+2]*W0[2*HID+c];
        Xbuf0[i] = (_Float16)(__sinf(30.0f * x) * Hm[(r & 31)*HID + c]);
    }
    __syncthreads();

    const int wave = tid >> 5;
    const int lane = tid & 31;
    const int rb = wave & 7;          // 16-row block
    const int cb = wave >> 3;         // 128-col half
    const int lh = lane & 15;
    const bool hi = lane >= 16;

    for (int layer = 1; layer <= 5; ++layer) {
        // stage modulation vector, bias and the full 128KB weight matrix in LDS
        const float* Hl = Hmod + (size_t)layer*(NB*NNODES*HID) + (size_t)b*(NNODES*HID);
        for (int i = tid; i < NNODES*HID; i += 512) Hm[i] = Hl[i];
        for (int i = tid; i < HID; i += 512) Bs[i] = BiasAll[layer*HID + i];
        {
            const uint4* wsrc = (const uint4*)(WT + (size_t)(layer - 1) * 65536);
            uint4* wdst = (uint4*)WL;
            #pragma unroll 4
            for (int i = tid; i < 8192; i += 512) wdst[i] = wsrc[i];   // 16B x 8192
        }
        __syncthreads();

        _Float16* Xc = (layer & 1) ? Xbuf0 : Xbuf1;   // read
        _Float16* Xn = (layer & 1) ? Xbuf1 : Xbuf0;   // write

        v8f acc[8] = {};
        #pragma unroll
        for (int kb = 0; kb < 8; ++kb) {
            // A fragment (ISA 16-bit A layout): lanes<16: K 0-7 & 16-23; lanes>=16: K 8-15 & 24-31
            const int arow = rb*16 + lh;
            const int koff = kb*32 + (hi ? 8 : 0);
            const v8h* ap = (const v8h*)(Xc + arow*HID + koff);
            v8h alo = ap[0];
            v8h ahi = ap[2];                          // +16 elements
            v16h a;
            #pragma unroll
            for (int i = 0; i < 8; ++i) { a[i] = alo[i]; a[i+8] = ahi[i]; }
            #pragma unroll
            for (int ct = 0; ct < 8; ++ct) {
                // B fragment from LDS: col n in lane%16; lanes<16 hold K 0-7 & 16-23, lanes>=16 hold K 8-15 & 24-31
                const int n = cb*128 + ct*16 + lh;
                const v8h* bp = (const v8h*)(WL + (size_t)n*HID + kb*32 + (hi ? 8 : 0));
                v8h blo = bp[0];
                v8h bhi = bp[2];
                v16h bf;
                #pragma unroll
                for (int i = 0; i < 8; ++i) { bf[i] = blo[i]; bf[i+8] = bhi[i]; }
                acc[ct] = __builtin_amdgcn_wmma_f32_16x16x32_f16(
                    false, a, false, bf, (short)0, acc[ct], false, false);
            }
        }
        // epilogue: sin(30x)*h, per documented C layout (lane%16 = N, VGPR v -> M = v + 8*(lane>=16))
        #pragma unroll
        for (int ct = 0; ct < 8; ++ct) {
            const int n = cb*128 + ct*16 + lh;
            #pragma unroll
            for (int v = 0; v < 8; ++v) {
                const int r = rb*16 + v + (hi ? 8 : 0);
                float x = acc[ct][v] + Bs[n];
                Xn[r*HID + n] = (_Float16)(__sinf(30.0f * x) * Hm[(r & 31)*HID + n]);
            }
        }
        __syncthreads();
    }

    // ---- layer 6 (dout=1) ----
    _Float16* Xf = Xbuf1;   // layer 5 wrote Xbuf1
    if (tid < 128) {
        float s = b6[0];
        for (int k = 0; k < HID; ++k) s += (float)Xf[tid*HID + k] * W6[k];
        sdf_out[row0 + tid] = s;
    }
}

// ------------------------------------------------------------------
// 7) RBF weights + weighted SDF
// ------------------------------------------------------------------
__global__ __launch_bounds__(256)
void k_rbf(const float* __restrict__ vis, const float* __restrict__ sdfs,
           const float* __restrict__ wsP, float* __restrict__ out_wsdf,
           float* __restrict__ out_pw, float* __restrict__ out_sd)
{
    int t = blockIdx.x * 256 + threadIdx.x;   // [b][ps], 8192 total
    int b = t >> 11;
    float wv[NNODES];
    float norm = 0.f;
    for (int n = 0; n < NNODES; ++n) {
        int bn = b * NNODES + n;
        const float* R  = wsP + 512  + bn * 9;
        const float* sc = wsP + 128  + bn * 3;
        const float* ce = wsP + 1664 + bn * 3;
        size_t o = ((size_t)t * NNODES + n) * 6;
        float d0 = vis[o+0]-ce[0], d1 = vis[o+1]-ce[1], d2 = vis[o+2]-ce[2];
        float l0 = (R[0]*d0 + R[3]*d1 + R[6]*d2) / sc[0];
        float l1 = (R[1]*d0 + R[4]*d1 + R[7]*d2) / sc[1];
        float l2 = (R[2]*d0 + R[5]*d1 + R[8]*d2) / sc[2];
        float scl = l0*l0 + l1*l1 + l2*l2;
        out_sd[(size_t)t * NNODES + n] = scl;
        float w = wsP[bn] * expf(-0.5f * scl);
        wv[n] = w;
        norm += w;
    }
    bool mask = (norm == 0.f);
    float inv = mask ? 0.f : 1.f / norm;
    float acc = 0.f;
    for (int n = 0; n < NNODES; ++n) {
        float pw = wv[n] * inv;
        out_pw[(size_t)t * NNODES + n] = pw;
        acc += pw * sdfs[(size_t)t * NNODES + n];
    }
    out_wsdf[t] = mask ? 1.0f : acc;
}

// ------------------------------------------------------------------
extern "C" void kernel_launch(void* const* d_in, const int* in_sizes, int n_in,
                              void* d_out, int out_size, void* d_ws, size_t ws_size,
                              hipStream_t stream)
{
    const float* pts  = (const float*)d_in[0];
    const float* extc = (const float*)d_in[1];
    const float* zc   = (const float*)d_in[2];
    const float* prob = (const float*)d_in[3];
    const float* odir = (const float*)d_in[4];
    const float* ou   = (const float*)d_in[5];
    const float* sW[7]; const float* sb[7];
    for (int i = 0; i < 7; ++i) { sW[i] = (const float*)d_in[6+i]; sb[i] = (const float*)d_in[13+i]; }
    ModPtrs mp; WPtrs wp;
    for (int i = 0; i < 6; ++i) { mp.W[i] = (const float*)d_in[20+i]; mp.bvec[i] = (const float*)d_in[26+i]; }
    for (int i = 0; i < 5; ++i) wp.W[i] = sW[1+i];
    for (int i = 0; i < 6; ++i) wp.bvec[i] = sb[i];

    float* out = (float*)d_out;
    char*  ws  = (char*)d_ws;
    float*    wsP     = (float*)(ws + WS_PARAMS);
    int*      idxsel  = (int*)(ws + WS_IDX);
    float*    Hmod    = (float*)(ws + WS_HMOD);
    _Float16* WT      = (_Float16*)(ws + WS_WT);
    float*    BiasAll = (float*)(ws + WS_BIAS);

    k_params<<<1, 128, 0, stream>>>(extc, wsP, out + O_EXT, out + O_CEN);
    k_prep<<<1280, 256, 0, stream>>>(wp, WT, BiasAll);
    k_mod<<<NB*NNODES, 256, 0, stream>>>(zc, mp, Hmod);
    k_topk<<<NB*NNODES, 1024, NPTS*sizeof(unsigned long long), stream>>>(pts, prob, wsP, idxsel);
    k_gather<<<MROWS/256, 256, 0, stream>>>(pts, odir, ou, idxsel, wsP,
                                            out + O_VIS, out + O_SEL, out + O_CI);
    k_siren<<<MROWS/128, 512, SM_TOTAL, stream>>>(out + O_CI, sW[0], WT, BiasAll,
                                                  Hmod, sW[6], sb[6], out + O_SDFS);
    k_rbf<<<NB*PSEL/256, 256, 0, stream>>>(out + O_VIS, out + O_SDFS, wsP,
                                           out + O_WSDF, out + O_PW, out + O_SD);
}
